// GATSelfAttention_6382321401929
// MI455X (gfx1250) — matrix operand order repaired and checked by reading, not compiled, not run on hardware
//
#include <hip/hip_runtime.h>
#include <hip/hip_bf16.h>

typedef __attribute__((ext_vector_type(16))) __bf16 v16bf;
typedef __attribute__((ext_vector_type(8)))  float  v8f;
typedef unsigned int u32x4 __attribute__((ext_vector_type(4)));
typedef int          i32x8 __attribute__((ext_vector_type(8)));
typedef int          i32x4 __attribute__((ext_vector_type(4)));

#define NTYPE 3
#define NB    32
#define EE    1024
#define IND   300
#define DDIM  128
#define QD    300
#define TWO_D 256
#define NEG_INF (-9.0e15f)

// ---- Tensor Data Mover availability -------------------------------------
#if __has_builtin(__builtin_amdgcn_tensor_load_to_lds)
#define HAVE_TDM 1
#if __has_include(<hip/amd_detail/amd_gfx1250_TDM.h>)
#define TDM_SIX_ARGS 1   // amdgpu-toolchain therock-10.0: 6-arg builtin
#else
#define TDM_SIX_ARGS 0   // ROCm 7.2 clang-22: 5-arg builtin (probe-verified)
#endif
#else
#define HAVE_TDM 0
#endif

#if HAVE_TDM
__device__ inline void wait_tensorcnt0() {
#if __has_builtin(__builtin_amdgcn_s_wait_tensorcnt)
  __builtin_amdgcn_s_wait_tensorcnt(0);
#else
  asm volatile("s_wait_tensorcnt 0x0" ::: "memory");
#endif
  __asm__ volatile("" ::: "memory");
}

// Issue a 2-D TDM tile load (global -> LDS), D# packed per CDNA5 ISA §8.3/8.4.
// dims/strides in elements of size (1<<dsz_log) bytes. LDS rows packed densely.
__device__ inline void tdm_load_2d(unsigned lds_addr, unsigned long long ga,
                                   unsigned dsz_log, unsigned tile0, unsigned tile1,
                                   unsigned tdim0, unsigned tdim1,
                                   unsigned long long stride0) {
  u32x4 g0; i32x8 g1; i32x4 g2, g3;
  g0[0] = 1u;                                       // count=1, user descriptor
  g0[1] = lds_addr;                                 // D#.lds_addr
  g0[2] = (unsigned)ga;                             // global_addr[31:0]
  g0[3] = (unsigned)((ga >> 32) & 0x01FFFFFFu) | (2u << 30);  // addr[56:32] | type=2
  g1[0] = (int)(dsz_log << 16);                     // wg_mask=0, data_size
  g1[1] = (int)((tdim0 & 0xFFFFu) << 16);           // tensor_dim0[15:0]
  g1[2] = (int)(((tdim0 >> 16) & 0xFFFFu) | ((tdim1 & 0xFFFFu) << 16));
  g1[3] = (int)(((tdim1 >> 16) & 0xFFFFu) | ((tile0 & 0xFFFFu) << 16));
  g1[4] = (int)(tile1 & 0xFFFFu);                   // tile_dim1, tile_dim2=0
  g1[5] = (int)(unsigned)(stride0 & 0xFFFFFFFFu);   // tensor_dim0_stride[31:0]
  g1[6] = (int)(unsigned)((stride0 >> 32) & 0xFFFFu);
  g1[7] = 0;
  g2[0] = g2[1] = g2[2] = g2[3] = 0;
  g3[0] = g3[1] = g3[2] = g3[3] = 0;
#if TDM_SIX_ARGS
  i32x8 g4; g4[0]=g4[1]=g4[2]=g4[3]=g4[4]=g4[5]=g4[6]=g4[7]=0;
  __builtin_amdgcn_tensor_load_to_lds(g0, g1, g2, g3, g4, 0);
#else
  __builtin_amdgcn_tensor_load_to_lds(g0, g1, g2, g3, 0);
#endif
}
#endif  // HAVE_TDM

// ---- WMMA fragment loaders (contiguous k = 16*half + e order in A and B, wave32) ----
// A (16x32 bf16, MxK): lane L: M = L&15, half = L>>4.
//   frag elems 0..15 = K = 16*half + 0..15    (contiguous)
__device__ inline v16bf frag_a(const __bf16* row, int hlf) {
  union { uint4 q[2]; v16bf v; } u;
  u.q[0] = *(const uint4*)(row + 16 * hlf);
  u.q[1] = *(const uint4*)(row + 16 * hlf + 8);
  return u.v;
}
// B (32x16 bf16, KxN): lane L: N = L&15, half = L>>4, K = 16*half + e (contiguous 16)
__device__ inline v16bf frag_b16(const __bf16* colbase) {
  union { uint4 q[2]; v16bf v; } u;
  u.q[0] = *(const uint4*)(colbase);
  u.q[1] = *(const uint4*)(colbase + 8);
  return u.v;
}
// B fragment gathered from a row-major [K][128] LDS tile at fixed column.
__device__ inline v16bf frag_b_gather(const __bf16* base) {
  v16bf v;
#pragma unroll
  for (int e = 0; e < 16; ++e) v[e] = base[(size_t)e * DDIM];
  return v;
}

// ---- K1: g[t][n][c] = sigmoid(relu(q @ qW1) @ qW2) ------------------------
__global__ __launch_bounds__(256) void gate_kernel(const float* __restrict__ q,
                                                   const float* __restrict__ qW1,
                                                   const float* __restrict__ qW2,
                                                   float* __restrict__ g) {
  int b = blockIdx.x;              // 0..95
  int t = b / NB, n = b % NB;
  int c = threadIdx.x;             // 0..255
  __shared__ float qs[QD];
  __shared__ float rs[TWO_D];
  for (int j = c; j < QD; j += 256) qs[j] = q[n * QD + j];
  __syncthreads();
  float acc = 0.f;
  const float* w1 = qW1 + (size_t)t * QD * TWO_D + c;
  for (int j = 0; j < QD; ++j) acc += qs[j] * w1[(size_t)j * TWO_D];
  rs[c] = acc > 0.f ? acc : 0.f;
  __syncthreads();
  float acc2 = 0.f;
  const float* w2 = qW2 + (size_t)t * TWO_D * TWO_D + c;
  for (int k = 0; k < TWO_D; ++k) acc2 += rs[k] * w2[(size_t)k * TWO_D];
  g[(t * NB + n) * TWO_D + c] = 1.f / (1.f + __expf(-acc2));
}

// ---- K2: v_left/v_right[t][n][k] = sum_d W[t][k][d] * g * a ---------------
__global__ __launch_bounds__(320) void vlr_kernel(const float* __restrict__ W,
                                                  const float* __restrict__ a,
                                                  const float* __restrict__ g,
                                                  float* __restrict__ vl,
                                                  float* __restrict__ vr) {
  int b = blockIdx.x; int t = b / NB, n = b % NB;
  __shared__ float gl[DDIM], gr[DDIM];
  int tid = threadIdx.x;
  if (tid < DDIM) {
    gl[tid] = g[(t * NB + n) * TWO_D + tid] * a[t * TWO_D + tid];
  } else if (tid < 2 * DDIM) {
    int d0 = tid - DDIM;
    gr[d0] = g[(t * NB + n) * TWO_D + DDIM + d0] * a[t * TWO_D + DDIM + d0];
  }
  __syncthreads();
  int k = tid;
  if (k < IND) {
    const float* wr = W + ((size_t)t * IND + k) * DDIM;
    float sl = 0.f, sr = 0.f;
    for (int d0 = 0; d0 < DDIM; ++d0) { float w = wr[d0]; sl += w * gl[d0]; sr += w * gr[d0]; }
    vl[(t * NB + n) * IND + k] = sl;
    vr[(t * NB + n) * IND + k] = sr;
  }
}

// ---- K3: left/right[t][n][e] = sum_k input[n][e][k] * v[t][n][k] ----------
__global__ __launch_bounds__(256) void lr_kernel(const float* __restrict__ x,
                                                 const float* __restrict__ vl,
                                                 const float* __restrict__ vr,
                                                 float* __restrict__ left,
                                                 float* __restrict__ right) {
  int R0 = blockIdx.x * 8;          // 8 rows per block (one per wave)
  int n = R0 >> 10;
  __shared__ float v6[6][IND];
  int tid = threadIdx.x;
  for (int idx = tid; idx < 6 * IND; idx += 256) {
    int t = idx / IND; int k = idx - t * IND;
    int tt = t % 3;
    v6[t][k] = (t < 3) ? vl[(tt * NB + n) * IND + k] : vr[(tt * NB + n) * IND + k];
  }
  __syncthreads();
  int w = tid >> 5, lane = tid & 31;
  int R = R0 + w;
  int e = R & 1023;
  float acc[6] = {0.f, 0.f, 0.f, 0.f, 0.f, 0.f};
  const float* xr = x + (size_t)R * IND;
  for (int k = lane; k < IND; k += 32) {
    float xv = xr[k];
#pragma unroll
    for (int t = 0; t < 6; ++t) acc[t] += xv * v6[t][k];
  }
#pragma unroll
  for (int t = 0; t < 6; ++t)
#pragma unroll
    for (int off = 16; off; off >>= 1) acc[t] += __shfl_xor(acc[t], off, 32);
  if (lane == 0) {
#pragma unroll
    for (int t = 0; t < 3; ++t) {
      left [(t * NB + n) * EE + e] = acc[t];
      right[(t * NB + n) * EE + e] = acc[3 + t];
    }
  }
}

// ---- K4: h(bf16) = input @ W[2]  via v_wmma_f32_16x16x32_bf16 -------------
__global__ __launch_bounds__(256) void hgemm_kernel(const float* __restrict__ x,
                                                    const float* __restrict__ W2,
                                                    __bf16* __restrict__ hbf) {
  __shared__ __bf16 Wt[DDIM][320];   // [dd][k], K padded 300->320 with zeros
  __shared__ __bf16 At[16][32];      // [e][k] bf16 A-tile
  int tid = threadIdx.x;
  int R0 = blockIdx.x * 16;
  for (int idx = tid; idx < 320 * DDIM; idx += 256) {
    int d0 = idx & (DDIM - 1);
    int k  = idx >> 7;
    float v = (k < IND) ? W2[(size_t)k * DDIM + d0] : 0.f;
    Wt[d0][k] = (__bf16)v;
  }
  int w = tid >> 5, lane = tid & 31;
  int hlf = lane >> 4, nn = lane & 15;
  int dd0 = w * 16;
  v8f acc = {};
  for (int kc = 0; kc < 320; kc += 32) {
    __syncthreads();
#pragma unroll
    for (int s = 0; s < 2; ++s) {
      int idx = tid + s * 256;
      int e = idx >> 5, k = idx & 31;
      int kg = kc + k;
      float v = (kg < IND) ? x[(size_t)(R0 + e) * IND + kg] : 0.f;
      At[e][k] = (__bf16)v;
    }
    __syncthreads();
    v16bf af = frag_a(&At[nn][0], hlf);
    v16bf bfr = frag_b16(&Wt[dd0 + nn][kc + 16 * hlf]);
    acc = __builtin_amdgcn_wmma_f32_16x16x32_bf16(false, af, false, bfr,
                                                  (short)0, acc, false, false);
  }
#pragma unroll
  for (int r = 0; r < 8; ++r) {
    int row = R0 + r + 8 * hlf;
    hbf[(size_t)row * DDIM + dd0 + nn] = (__bf16)acc[r];
  }
}

// ---- K5: per-row online softmax stats (max, 1/denom) over j ---------------
__global__ __launch_bounds__(256) void rowstat_kernel(const int* __restrict__ adj,
                                                      const float* __restrict__ left,
                                                      const float* __restrict__ right,
                                                      float* __restrict__ mrow,
                                                      float* __restrict__ zrow) {
  int blk = blockIdx.x;                 // 4096
  int n = blk >> 7;
  int i0 = (blk & 127) * 8;
  __shared__ float rt[3][EE];
  int tid = threadIdx.x;
#if HAVE_TDM
  if (tid < 32) {
    // One DMA: 3 rows of 1024 f32, row stride NB*EE, packed into rt[3][1024].
    tdm_load_2d((unsigned)(size_t)&rt[0][0],
                (unsigned long long)(size_t)(right + (size_t)n * EE),
                2u, EE, 3u, EE, 3u, (unsigned long long)(NB * EE));
    wait_tensorcnt0();
  }
#else
  for (int idx = tid; idx < 3 * EE; idx += 256) {
    int t = idx >> 10, j = idx & 1023;
    rt[t][j] = right[(t * NB + n) * EE + j];
  }
#endif
  __syncthreads();
  int w = tid >> 5, lane = tid & 31;
  int i = i0 + w;
  float l0 = left[(0 * NB + n) * EE + i];
  float l1 = left[(1 * NB + n) * EE + i];
  float l2 = left[(2 * NB + n) * EE + i];
  const int* arow = adj + ((size_t)n << 20) + (size_t)i * EE;
  float m = NEG_INF, sum = 0.f;
  for (int j = lane; j < EE; j += 32) {
    int a = arow[j];
    float s;
    if (a > 0) {
      float lv = (a == 1) ? l0 : (a == 2 ? l1 : l2);
      s = lv + rt[a - 1][j];
      s = s > 0.f ? s : 0.2f * s;
    } else s = NEG_INF;
    if (s > m) { sum = sum * __expf(m - s) + 1.f; m = s; }
    else sum += __expf(s - m);
  }
#pragma unroll
  for (int off = 16; off; off >>= 1) {
    float m2 = __shfl_xor(m, off, 32);
    float s2 = __shfl_xor(sum, off, 32);
    float mn = m > m2 ? m : m2;
    sum = sum * __expf(m - mn) + s2 * __expf(m2 - mn);
    m = mn;
  }
  if (lane == 0) { mrow[n * EE + i] = m; zrow[n * EE + i] = 1.f / sum; }
}

// ---- K6: out[n,j,:] = sum_i softmax(score)[i,j] * h[i,:]  (fused, WMMA) ---
__global__ __launch_bounds__(256) void out_kernel(const int* __restrict__ adj,
                                                  const float* __restrict__ left,
                                                  const float* __restrict__ mrow,
                                                  const float* __restrict__ zrow,
                                                  const float* __restrict__ right,
                                                  const __bf16* __restrict__ hbf,
                                                  float* __restrict__ out) {
  int blk = blockIdx.x;                 // 2048 = 32 n * 64 j-tiles
  int n = blk >> 6;
  int j0 = (blk & 63) * 16;
  __shared__ __bf16 cT[16 * 32];        // A-tile: coef^T  [jj][ii], stride 32
#if HAVE_TDM
  __shared__ __bf16 hRM[32 * DDIM];     // B-tile staged row-major by the TDM
#else
  __shared__ __bf16 hT[DDIM * 48];      // B-tile: h^T [dd][ii], stride 48 (32B chunks)
#endif
  __shared__ float rjs[3][16];
  int tid = threadIdx.x;
  if (tid < 48) {
    int t = tid >> 4, jj = tid & 15;
    rjs[t][jj] = right[(t * NB + n) * EE + j0 + jj];
  }
  int w = tid >> 5, lane = tid & 31;
  int hlf = lane >> 4, nn = lane & 15;
  int dd0 = w * 16;
  v8f acc = {};
#if !HAVE_TDM
  const uint4* hq = (const uint4*)hbf;  // 8 bf16 per uint4, 16 uint4 per h row
#endif
  for (int i0 = 0; i0 < EE; i0 += 32) {
    __syncthreads();
#if HAVE_TDM
    // Kick the h-tile DMA first so it overlaps the softmax VALU work below.
    if (tid < 32)
      tdm_load_2d((unsigned)(size_t)hRM,
                  (unsigned long long)(size_t)(hbf + (size_t)(n * EE + i0) * DDIM),
                  1u, DDIM, 32u, DDIM, 32u, (unsigned long long)DDIM);
#endif
    // prefetch next adj tile while we work on this one
    if (i0 + 32 < EE)
      __builtin_prefetch(adj + ((size_t)n << 20) + (size_t)(i0 + 32 + (tid >> 4)) * EE + j0, 0, 1);
    // reconstruct softmax coefficients for this 32i x 16j tile, transposed
#pragma unroll
    for (int s = 0; s < 2; ++s) {
      int e = tid + s * 256;
      int ii = e >> 4, jj = e & 15;
      int ig = i0 + ii;
      int a = adj[((size_t)n << 20) + (size_t)ig * EE + j0 + jj];
      int t = a > 0 ? a - 1 : 0;
      float sc;
      if (a > 0) {
        float s1 = left[(t * NB + n) * EE + ig] + rjs[t][jj];
        sc = s1 > 0.f ? s1 : 0.2f * s1;
      } else sc = NEG_INF;
      float c = __expf(sc - mrow[n * EE + ig]) * zrow[n * EE + ig];
      cT[jj * 32 + ii] = (__bf16)c;
    }
#if HAVE_TDM
    if (tid < 32) wait_tensorcnt0();
#else
    // stage h tile (32 i-rows x 128 dd) K-contiguous (transposed) in LDS
    {
      int ii = tid >> 3;
      int seg = tid & 7;
      size_t base = ((size_t)(n * EE) + i0 + ii) * (DDIM / 8) + seg * 2;
      union { uint4 q[2]; __bf16 u[16]; } hu;
      hu.q[0] = hq[base];
      hu.q[1] = hq[base + 1];
#pragma unroll
      for (int c0 = 0; c0 < 16; ++c0)
        hT[(seg * 16 + c0) * 48 + ii] = hu.u[c0];
    }
#endif
    __syncthreads();
    v16bf af  = frag_a(&cT[nn * 32], hlf);
#if HAVE_TDM
    v16bf bfr = frag_b_gather(&hRM[(16 * hlf) * DDIM + dd0 + nn]);
#else
    v16bf bfr = frag_b16(&hT[(dd0 + nn) * 48 + 16 * hlf]);
#endif
    acc = __builtin_amdgcn_wmma_f32_16x16x32_bf16(false, af, false, bfr,
                                                  (short)0, acc, false, false);
  }
#pragma unroll
  for (int r = 0; r < 8; ++r) {
    int j = j0 + r + 8 * hlf;
    out[((size_t)(n * EE + j)) * DDIM + dd0 + nn] = acc[r];
  }
}

extern "C" void kernel_launch(void* const* d_in, const int* in_sizes, int n_in,
                              void* d_out, int out_size, void* d_ws, size_t ws_size,
                              hipStream_t stream) {
  (void)in_sizes; (void)n_in; (void)out_size; (void)ws_size;
  const float* x   = (const float*)d_in[0];
  const int*   adj = (const int*)d_in[1];
  // d_in[2] = entity_mask: unused by the reference
  const float* qv  = (const float*)d_in[3];
  const float* W   = (const float*)d_in[4];
  const float* a   = (const float*)d_in[5];
  const float* qW1 = (const float*)d_in[6];
  const float* qW2 = (const float*)d_in[7];
  float* out = (float*)d_out;

  char* ws = (char*)d_ws;
  float* g     = (float*)ws; ws += (size_t)NTYPE * NB * TWO_D * 4;
  float* vl    = (float*)ws; ws += (size_t)NTYPE * NB * IND * 4;
  float* vr    = (float*)ws; ws += (size_t)NTYPE * NB * IND * 4;
  float* left  = (float*)ws; ws += (size_t)NTYPE * NB * EE * 4;
  float* right = (float*)ws; ws += (size_t)NTYPE * NB * EE * 4;
  float* mrow  = (float*)ws; ws += (size_t)NB * EE * 4;
  float* zrow  = (float*)ws; ws += (size_t)NB * EE * 4;
  __bf16* hbf  = (__bf16*)ws;                  // 32*1024*128 bf16 = 8 MB

  gate_kernel  <<<NTYPE * NB, 256, 0, stream>>>(qv, qW1, qW2, g);
  vlr_kernel   <<<NTYPE * NB, 320, 0, stream>>>(W, a, g, vl, vr);
  lr_kernel    <<<(NB * EE) / 8, 256, 0, stream>>>(x, vl, vr, left, right);
  hgemm_kernel <<<(NB * EE) / 16, 256, 0, stream>>>(x, W + (size_t)2 * IND * DDIM, hbf);
  rowstat_kernel<<<(NB * EE) / 8, 256, 0, stream>>>(adj, left, right, mrow, zrow);
  out_kernel   <<<NB * (EE / 16), 256, 0, stream>>>(adj, left, mrow, zrow, right, hbf, out);
}
